// AttentionGlobalPooling_7258494730679
// MI455X (gfx1250) — compile-verified, matrix operand order checked by reading, not run on hardware
//
#include <hip/hip_runtime.h>
#include <hip/hip_bf16.h>

typedef __attribute__((ext_vector_type(16))) __bf16 v16bf;
typedef __attribute__((ext_vector_type(8)))  float  v8f;

#define IN_CH 128
#define HID_CH 128
#define WSTR 144          // LDS row stride (bf16 elems): 288B rows -> 32B-aligned fragments
#define NPB 512           // nodes per block in pooling pass

// ---------------------------------------------------------------------------
// Pass 1: per-node score = (relu(x @ W1 + b1) @ W2 + b2)
// 256 threads = 8 waves; each wave computes a 16-row tile with WMMA bf16.
// ---------------------------------------------------------------------------
__global__ __launch_bounds__(256) void k_scores(
    const float* __restrict__ x, const float* __restrict__ W1,
    const float* __restrict__ b1, const float* __restrict__ W2,
    const float* __restrict__ b2, float* __restrict__ scores, int N)
{
    // W1 transposed into LDS as bf16: sW[n*WSTR + k]  (n = hidden col, k = input row)
    __shared__ __attribute__((aligned(32))) __bf16 sW[HID_CH * WSTR];
    const int tid = threadIdx.x;
    for (int idx = tid; idx < IN_CH * HID_CH; idx += 256) {
        const int k = idx >> 7;        // W1 is [k][n] row-major
        const int n = idx & 127;
        sW[n * WSTR + k] = (__bf16)W1[idx];
    }
    __syncthreads();

    const int wave = tid >> 5;
    const int lane = tid & 31;
    const int l16  = lane & 15;
    const int hi   = lane >> 4;                    // K-half select (wave32 layout)
    const int rowBase = blockIdx.x * 128 + wave * 16;
    if (rowBase >= N) return;                      // wave-uniform exit (EXEC stays all-1)

    int row = rowBase + l16;
    if (row >= N) row = N - 1;                     // clamp loads; stores masked later
    const float* __restrict__ xrow = x + (size_t)row * IN_CH;

    const v8f vzero = {0.f,0.f,0.f,0.f,0.f,0.f,0.f,0.f};
    v8f acc[8];
#pragma unroll
    for (int t = 0; t < 8; ++t) acc[t] = vzero;

#pragma unroll
    for (int kst = 0; kst < 4; ++kst) {
        const int kOff = kst * 32 + hi * 16;
        // A fragment: lane holds row (l16), K = kOff..kOff+15, packed 2/VGPR
        v16bf a;
#pragma unroll
        for (int i = 0; i < 16; ++i) a[i] = (__bf16)xrow[kOff + i];
#pragma unroll
        for (int t = 0; t < 8; ++t) {
            // B fragment: lane holds column N = t*16+l16, same K range
            const v16bf bfrag = *(const v16bf*)&sW[(t * 16 + l16) * WSTR + kOff];
            acc[t] = __builtin_amdgcn_wmma_f32_16x16x32_bf16(
                false, a, false, bfrag, (short)0, acc[t], false, false);
        }
    }

    // bias + ReLU + dot with W2 directly on the C-layout registers:
    // acc[t] VGPR r, lanes 0-15 -> (M=r, Ncol=t*16+l16); lanes 16-31 -> (M=r+8, same col)
    float srow[8];
#pragma unroll
    for (int r = 0; r < 8; ++r) srow[r] = 0.f;
#pragma unroll
    for (int t = 0; t < 8; ++t) {
        const int n = t * 16 + l16;
        const float bias = b1[n];
        const float w2   = W2[n];
#pragma unroll
        for (int r = 0; r < 8; ++r) {
            float hv = acc[t][r] + bias;
            hv = fmaxf(hv, 0.f);
            srow[r] = fmaf(hv, w2, srow[r]);
        }
    }
    // reduce across the 16 lanes of each half (columns) — wave32 width-16 butterflies
#pragma unroll
    for (int m = 8; m >= 1; m >>= 1) {
#pragma unroll
        for (int r = 0; r < 8; ++r) srow[r] += __shfl_xor(srow[r], m, 16);
    }
    if (l16 == 0) {
        const float bb2 = b2[0];
#pragma unroll
        for (int r = 0; r < 8; ++r) {
            const int m = rowBase + r + hi * 8;
            if (m < N) scores[m] = srow[r] + bb2;
        }
    }
}

// ---------------------------------------------------------------------------
// Pass 2: init output + segment accumulators (d_out / d_ws are poisoned)
// ---------------------------------------------------------------------------
__global__ void k_init(float* __restrict__ out, float* __restrict__ gmax,
                       float* __restrict__ gsum, int G)
{
    const int i = blockIdx.x * blockDim.x + threadIdx.x;
    if (i < G * IN_CH) out[i] = 0.f;
    if (i < G) { gmax[i] = -__builtin_inff(); gsum[i] = 0.f; }
}

// ---------------------------------------------------------------------------
// Pass 3: segment max (batch is sorted -> run-compress atomics)
// ---------------------------------------------------------------------------
__global__ void k_segmax(const float* __restrict__ scores,
                         const long long* __restrict__ batch,
                         float* __restrict__ gmax, int N)
{
    const int t = blockIdx.x * blockDim.x + threadIdx.x;
    const int start = t * 8;
    if (start >= N) return;
    const int end = min(start + 8, N);
    int curg = (int)batch[start];
    float m = scores[start];
    for (int n = start + 1; n < end; ++n) {
        const int g = (int)batch[n];
        const float s = scores[n];
        if (g != curg) {
            __hip_atomic_fetch_max(&gmax[curg], m, __ATOMIC_RELAXED, __HIP_MEMORY_SCOPE_AGENT);
            curg = g; m = s;
        } else {
            m = fmaxf(m, s);
        }
    }
    __hip_atomic_fetch_max(&gmax[curg], m, __ATOMIC_RELAXED, __HIP_MEMORY_SCOPE_AGENT);
}

// ---------------------------------------------------------------------------
// Pass 4: e = exp(score - segmax), segment sum of e
// ---------------------------------------------------------------------------
__global__ void k_expsum(const float* __restrict__ scores,
                         const long long* __restrict__ batch,
                         const float* __restrict__ gmax,
                         float* __restrict__ ebuf, float* __restrict__ gsum, int N)
{
    const int t = blockIdx.x * blockDim.x + threadIdx.x;
    const int start = t * 8;
    if (start >= N) return;
    const int end = min(start + 8, N);
    int curg = (int)batch[start];
    float acc = 0.f;
    for (int n = start; n < end; ++n) {
        const int g = (int)batch[n];
        if (g != curg) { atomicAdd(&gsum[curg], acc); acc = 0.f; curg = g; }
        const float e = __expf(scores[n] - gmax[g]);
        ebuf[n] = e;
        acc += e;
    }
    atomicAdd(&gsum[curg], acc);
}

// ---------------------------------------------------------------------------
// Pass 5: attn = e / (segsum + 1e-16), in place
// ---------------------------------------------------------------------------
__global__ void k_norm(float* __restrict__ ebuf,
                       const long long* __restrict__ batch,
                       const float* __restrict__ gsum, int N)
{
    const int n = blockIdx.x * blockDim.x + threadIdx.x;
    if (n < N) {
        const int g = (int)batch[n];
        ebuf[n] = ebuf[n] / (gsum[g] + 1e-16f);
    }
}

// ---------------------------------------------------------------------------
// Pass 6: out[g][c] += x[n][c] * attn[n]  (full f32 x; sorted -> boundary atomics)
// 128 threads = one channel each; 512 nodes per block staged through LDS.
// ---------------------------------------------------------------------------
__global__ __launch_bounds__(128) void k_pool(
    const float* __restrict__ x, const float* __restrict__ attn,
    const long long* __restrict__ batch, float* __restrict__ out, int N)
{
    __shared__ float sA[NPB];
    __shared__ int   sG[NPB];
    const int start = blockIdx.x * NPB;
    if (start >= N) return;
    const int cnt = min(NPB, N - start);
    for (int i = threadIdx.x; i < cnt; i += 128) {
        sA[i] = attn[start + i];
        sG[i] = (int)batch[start + i];
    }
    __syncthreads();

    const int c = threadIdx.x;   // channel 0..127, coalesced across the block
    float acc = 0.f;
    int curg = sG[0];
    for (int i = 0; i < cnt; ++i) {
        const int g = sG[i];
        if (g != curg) {
            atomicAdd(&out[(size_t)curg * IN_CH + c], acc);
            acc = 0.f; curg = g;
        }
        const size_t n = (size_t)(start + i);
        if (((i & 7) == 0) && (i + 16 < cnt))
            __builtin_prefetch(x + (n + 16) * IN_CH + c, 0, 0);  // global_prefetch_b8
        acc = fmaf(x[n * IN_CH + c], sA[i], acc);
    }
    atomicAdd(&out[(size_t)curg * IN_CH + c], acc);
}

// ---------------------------------------------------------------------------
extern "C" void kernel_launch(void* const* d_in, const int* in_sizes, int n_in,
                              void* d_out, int out_size, void* d_ws, size_t ws_size,
                              hipStream_t stream)
{
    const float*     x     = (const float*)d_in[0];
    const float*     W1    = (const float*)d_in[1];
    const float*     b1    = (const float*)d_in[2];
    const float*     W2    = (const float*)d_in[3];
    const float*     b2    = (const float*)d_in[4];
    const long long* batch = (const long long*)d_in[5];
    float* out = (float*)d_out;

    const int N = in_sizes[0] / IN_CH;     // 500,000
    const int G = out_size / IN_CH;        // 4,096

    // workspace layout: scores[N] | attn[N] | gmax[G] | gsum[G]
    float* scores = (float*)d_ws;
    float* attn   = scores + N;
    float* gmax   = attn + N;
    float* gsum   = gmax + G;

    k_init  <<<(G * IN_CH + 255) / 256, 256, 0, stream>>>(out, gmax, gsum, G);
    k_scores<<<(N + 127) / 128,        256, 0, stream>>>(x, W1, b1, W2, b2, scores, N);
    k_segmax<<<((N + 7) / 8 + 255) / 256, 256, 0, stream>>>(scores, batch, gmax, N);
    k_expsum<<<((N + 7) / 8 + 255) / 256, 256, 0, stream>>>(scores, batch, gmax, attn, gsum, N);
    k_norm  <<<(N + 255) / 256,        256, 0, stream>>>(attn, batch, gsum, N);
    k_pool  <<<(N + NPB - 1) / NPB,    128, 0, stream>>>(x, attn, batch, out, N);
}
